// NREVAttention_1657857376276
// MI455X (gfx1250) — matrix operand order repaired and checked by reading, not compiled, not run on hardware
//
#include <hip/hip_runtime.h>
#include <math.h>

typedef __attribute__((ext_vector_type(16))) _Float16 v16h;
typedef __attribute__((ext_vector_type(8)))  _Float16 v8h;
typedef __attribute__((ext_vector_type(2)))  _Float16 h2;
typedef __attribute__((ext_vector_type(8)))  float    v8f;

// Problem constants (reference: B=2, H=16, L=2048, D=64)
#define NB 2
#define NH 16
#define NL 2048
#define ND 64
#define QT 16             // query rows per wave
#define KT 64             // key rows per (double-buffered) LDS tile
#define WAVES 8           // waves per block
#define QBLK (WAVES * QT) // 128 query rows per block
#define NQB (NL / QBLK)   // 16 query blocks per (b,h)
#define NT (NL / KT)      // 32 key tiles
#define NW64 (NL / 64)    // 32 packed mask words per row

// log2(e) folded constants: softmax computed in base-2 (v_exp_f32 is 2^x).
#define QSCALE   0.18033688011112042f   /* 0.125 * log2(e) */
#define MASKNEG  -14426.950408889634f   /* -10000 * log2(e) */

// ---------------------------------------------------------------------------
// Pre-pass: bit-pack mask [B,1,L,L] int32 -> [B*L][L/64] uint64 via __ballot.
// Cuts mask read traffic in the attention kernel by 32x.
// ---------------------------------------------------------------------------
__global__ __launch_bounds__(256)
void mask_pack_kernel(const int* __restrict__ mask,
                      unsigned long long* __restrict__ mpk)
{
    const int lane = threadIdx.x & 31;
    const int row  = blockIdx.x * (blockDim.x >> 5) + (threadIdx.x >> 5); // b*NL + q
    const int* src = mask + (size_t)row * NL;
    unsigned long long* dst = mpk + (size_t)row * NW64;
    for (int wi = 0; wi < NW64; ++wi) {
        const int m0 = src[wi * 64 + lane];
        const int m1 = src[wi * 64 + 32 + lane];
        const unsigned long long b0 = __ballot(m0 != 0);
        const unsigned long long b1 = __ballot(m1 != 0);
        if (lane == 0) dst[wi] = (b0 & 0xffffffffull) | (b1 << 32);
    }
}

// ---------------------------------------------------------------------------
// Flash attention with NegativeReLU(V), cooperative f16 staging, WMMA core.
// ---------------------------------------------------------------------------
__global__ __launch_bounds__(256)
void nrev_flash_attn_kernel(const float* __restrict__ q,
                            const float* __restrict__ k,
                            const float* __restrict__ v,
                            const unsigned long long* __restrict__ mpk,
                            float*       __restrict__ out)
{
    __shared__ _Float16 k16[2][KT][ND];      // K tile, row-major f16       (16 KB)
    __shared__ _Float16 vt16[2][ND][KT];     // V tile, transposed, negReLU (16 KB)
    __shared__ _Float16 pbuf[WAVES][QT][KT]; // P row-major per wave        (16 KB)

    const int tid  = threadIdx.x;
    const int lane = tid & 31;
    const int wib  = tid >> 5;
    const int hi   = lane >> 4;
    const int lo   = lane & 15;

    const int bid = blockIdx.x;
    const int qb  = bid & (NQB - 1);
    const int h   = (bid / NQB) & (NH - 1);
    const int b   = bid / (NQB * NH);

    const float* __restrict__ qh = q + ((size_t)(b * NH + h) * NL) * ND;
    const float* __restrict__ kh = k + ((size_t)(b * NH + h) * NL) * ND;
    const float* __restrict__ vh = v + ((size_t)(b * NH + h) * NL) * ND;
    const unsigned long long* __restrict__ mpb = mpk + (size_t)b * NL * NW64;
    float* __restrict__ oh = out + ((size_t)(b * NH + h) * NL) * ND;

    const int q0 = qb * QBLK + wib * QT;

    // ---- Q tile (16x64) as two A fragments (f16, k=32 each), pre-scaled to log2 domain.
    // A k order used here (contiguous, B matches): lane row m = lane&15; element i -> k = hi*16 + i
    v16h qa[2];
#pragma unroll
    for (int c = 0; c < 2; ++c) {
#pragma unroll
        for (int g = 0; g < 2; ++g) {
            const float* srcq = qh + (size_t)(q0 + lo) * ND + c * 32 + hi * 16 + g * 8;
            float4 f0 = *(const float4*)(srcq + 0);
            float4 f1 = *(const float4*)(srcq + 4);
            qa[c][g * 8 + 0] = (_Float16)(f0.x * QSCALE);
            qa[c][g * 8 + 1] = (_Float16)(f0.y * QSCALE);
            qa[c][g * 8 + 2] = (_Float16)(f0.z * QSCALE);
            qa[c][g * 8 + 3] = (_Float16)(f0.w * QSCALE);
            qa[c][g * 8 + 4] = (_Float16)(f1.x * QSCALE);
            qa[c][g * 8 + 5] = (_Float16)(f1.y * QSCALE);
            qa[c][g * 8 + 6] = (_Float16)(f1.z * QSCALE);
            qa[c][g * 8 + 7] = (_Float16)(f1.w * QSCALE);
        }
    }

    // ---- Cooperative stage: K row-major f16, V transposed f16 with negReLU.
    auto stage = [&](int kbs, int bufi) {
        const int row = tid >> 2;          // 0..63
        const int cb  = (tid & 3) << 4;    // 0,16,32,48
        {
            const float* src = kh + (size_t)(kbs + row) * ND + cb;
            v16h t;
#pragma unroll
            for (int j = 0; j < 16; j += 4) {
                float4 f = *(const float4*)(src + j);
                t[j + 0] = (_Float16)f.x;
                t[j + 1] = (_Float16)f.y;
                t[j + 2] = (_Float16)f.z;
                t[j + 3] = (_Float16)f.w;
            }
            *(v16h*)&k16[bufi][row][cb] = t;   // 2x ds_store_b128
            if (kbs + KT < NL) __builtin_prefetch(src + (size_t)KT * ND, 0, 1);
        }
        {
            const float* src = vh + (size_t)(kbs + row) * ND + cb;
#pragma unroll
            for (int j = 0; j < 16; j += 4) {
                float4 f = *(const float4*)(src + j);
                vt16[bufi][cb + j + 0][row] = (_Float16)fminf(f.x, 0.0f);
                vt16[bufi][cb + j + 1][row] = (_Float16)fminf(f.y, 0.0f);
                vt16[bufi][cb + j + 2][row] = (_Float16)fminf(f.z, 0.0f);
                vt16[bufi][cb + j + 3][row] = (_Float16)fminf(f.w, 0.0f);
            }
            if (kbs + KT < NL) __builtin_prefetch(src + (size_t)KT * ND, 0, 1);
        }
    };

    // ---- State: O accumulators (4 C tiles), row max (log2 domain), per-lane
    //      partial row sums (cross-lane reduction deferred to epilogue).
    v8f acc[4];
    float mrow[8], lrow[8];
#pragma unroll
    for (int r = 0; r < 8; ++r) { mrow[r] = -INFINITY; lrow[r] = 0.0f; }
#pragma unroll
    for (int dt = 0; dt < 4; ++dt)
#pragma unroll
        for (int r = 0; r < 8; ++r) acc[dt][r] = 0.0f;

    stage(0, 0);
    __syncthreads();

    for (int kt = 0; kt < NT; ++kt) {
        const int p  = kt & 1;
        const int kb = kt * KT;

        if (kt + 1 < NT) stage(kb + KT, 1 - p);

        // ---- S = (Q*scale) K^T : four 16x16 key tiles, k-dim 64 (2 chunks).
        v8f s[4];
#pragma unroll
        for (int t = 0; t < 4; ++t) {
            v8f sa;
#pragma unroll
            for (int r = 0; r < 8; ++r) sa[r] = 0.0f;
#pragma unroll
            for (int c = 0; c < 2; ++c) {
                v16h kf = *(const v16h*)&k16[p][t * 16 + lo][c * 32 + hi * 16];
                sa = __builtin_amdgcn_wmma_f32_16x16x32_f16(
                        false, qa[c], false, kf, (short)0, sa, false, false);
            }
            s[t] = sa;
        }

        // ---- Apply bit-packed mask in place + per-lane row max.
        float cml[8];
#pragma unroll
        for (int r = 0; r < 8; ++r) {
            const int qrow = q0 + r + 8 * hi;
            const unsigned long long w = mpb[(size_t)qrow * NW64 + kt];
            const unsigned int w0 = (unsigned int)w;
            const unsigned int w1 = (unsigned int)(w >> 32);
            s[0][r] = ((w0 >> lo) & 1u)        ? s[0][r] : MASKNEG;
            s[1][r] = ((w0 >> (16 + lo)) & 1u) ? s[1][r] : MASKNEG;
            s[2][r] = ((w1 >> lo) & 1u)        ? s[2][r] : MASKNEG;
            s[3][r] = ((w1 >> (16 + lo)) & 1u) ? s[3][r] : MASKNEG;
            cml[r] = fmaxf(fmaxf(s[0][r], s[1][r]), fmaxf(s[2][r], s[3][r]));
        }

        // ---- Cross-lane row max: two rows packed per dword (v_pk_max_num_f16).
        h2 cmp2[4];
#pragma unroll
        for (int j = 0; j < 4; ++j) {
            h2 t; t[0] = (_Float16)cml[2 * j]; t[1] = (_Float16)cml[2 * j + 1];
            cmp2[j] = t;
        }
#pragma unroll
        for (int off = 1; off <= 8; off <<= 1) {
#pragma unroll
            for (int j = 0; j < 4; ++j) {
                int xi = __shfl_xor(__builtin_bit_cast(int, cmp2[j]), off, 32);
                cmp2[j] = __builtin_elementwise_max(cmp2[j], __builtin_bit_cast(h2, xi));
            }
        }

        // ---- Rescale only when a row max actually increased (rare after warmup).
        float mn[8];
        bool up = false;
#pragma unroll
        for (int r = 0; r < 8; ++r) {
            const float cmf = (float)cmp2[r >> 1][r & 1];
            mn[r] = fmaxf(mrow[r], cmf);
            up = up || (mn[r] > mrow[r]);
        }
        if (__any(up)) {
#pragma unroll
            for (int r = 0; r < 8; ++r) {
                const float cr = exp2f(mrow[r] - mn[r]);
                mrow[r] = mn[r];
                lrow[r] *= cr;
#pragma unroll
                for (int dt = 0; dt < 4; ++dt) acc[dt][r] *= cr;
            }
        }

        // ---- P = 2^(s - m); write straight to per-wave LDS slab; local row sums.
#pragma unroll
        for (int r = 0; r < 8; ++r) {
            const float base = mrow[r];
            const float e0 = exp2f(s[0][r] - base);
            const float e1 = exp2f(s[1][r] - base);
            const float e2 = exp2f(s[2][r] - base);
            const float e3 = exp2f(s[3][r] - base);
            const int row = r + 8 * hi;
            pbuf[wib][row][lo]      = (_Float16)e0;
            pbuf[wib][row][16 + lo] = (_Float16)e1;
            pbuf[wib][row][32 + lo] = (_Float16)e2;
            pbuf[wib][row][48 + lo] = (_Float16)e3;
            lrow[r] += (e0 + e1) + (e2 + e3);
        }

        asm volatile("s_wait_dscnt 0" ::: "memory");  // in-wave P store->load

        // ---- P A fragments (two k=32 chunks), 16B LDS loads.
        v16h pa[2];
#pragma unroll
        for (int cc = 0; cc < 2; ++cc) {
            v8h a0 = *(const v8h*)&pbuf[wib][lo][cc * 32 + hi * 16];
            v8h a1 = *(const v8h*)&pbuf[wib][lo][cc * 32 + hi * 16 + 8];
            pa[cc] = __builtin_shufflevector(a0, a1, 0, 1, 2, 3, 4, 5, 6, 7,
                                                     8, 9, 10, 11, 12, 13, 14, 15);
        }

        // ---- O += P * V_neg : four d tiles x two key chunks = 8 WMMAs.
#pragma unroll
        for (int dt = 0; dt < 4; ++dt) {
#pragma unroll
            for (int cc = 0; cc < 2; ++cc) {
                v16h vf = *(const v16h*)&vt16[p][dt * 16 + lo][cc * 32 + hi * 16];
                acc[dt] = __builtin_amdgcn_wmma_f32_16x16x32_f16(
                            false, pa[cc], false, vf, (short)0, acc[dt], false, false);
            }
        }

        __syncthreads();
    }

    // ---- Epilogue: one deferred cross-lane sum of l, then O / l.
#pragma unroll
    for (int r = 0; r < 8; ++r) {
#pragma unroll
        for (int off = 1; off <= 8; off <<= 1)
            lrow[r] += __shfl_xor(lrow[r], off, 32);
    }
#pragma unroll
    for (int dt = 0; dt < 4; ++dt) {
#pragma unroll
        for (int r = 0; r < 8; ++r) {
            const int qrow = q0 + r + 8 * hi;
            oh[(size_t)qrow * ND + dt * 16 + lo] = acc[dt][r] / lrow[r];
        }
    }
}

extern "C" void kernel_launch(void* const* d_in, const int* in_sizes, int n_in,
                              void* d_out, int out_size, void* d_ws, size_t ws_size,
                              hipStream_t stream) {
    (void)in_sizes; (void)n_in; (void)out_size; (void)ws_size;
    const float* q    = (const float*)d_in[0];
    const float* k    = (const float*)d_in[1];
    const float* v    = (const float*)d_in[2];
    const int*   mask = (const int*)d_in[3];
    float*       out  = (float*)d_out;

    // Bit-packed mask lives in workspace: NB*NL*NW64*8 = 1 MB.
    unsigned long long* mpk = (unsigned long long*)d_ws;

    mask_pack_kernel<<<(NB * NL) / WAVES, WAVES * 32, 0, stream>>>(mask, mpk);

    const int blocks = NB * NH * NQB;   // 512 blocks x 256 threads (8 waves)
    nrev_flash_attn_kernel<<<blocks, WAVES * 32, 0, stream>>>(q, k, v, mpk, out);
}
